// VJEPA2Predictor_47528108098033
// MI455X (gfx1250) — matrix operand order repaired and checked by reading, not compiled, not run on hardware
//
#include <hip/hip_runtime.h>
#include <cstdint>
#include <cstddef>

// ---------------------------------------------------------------------------
// Types for CDNA5 WMMA (gfx1250, wave32)
// ---------------------------------------------------------------------------
typedef __attribute__((ext_vector_type(8)))  float   v8f;
typedef __attribute__((ext_vector_type(16))) __bf16  v16bf;
typedef __attribute__((ext_vector_type(2)))  __bf16  v2bf;

typedef union {
    v16bf        v;
    unsigned int u[8];
} Frag;

__device__ __forceinline__ unsigned short f2bf(float f) {
    return __builtin_bit_cast(unsigned short, (__bf16)f);   // hw v_cvt, RTNE
}
// build_vector(fptrunc,fptrunc) -> single v_cvt_pk_bf16_f32
__device__ __forceinline__ unsigned int pack2(float lo, float hi) {
    v2bf t;
    t.x = (__bf16)lo;
    t.y = (__bf16)hi;
    return __builtin_bit_cast(unsigned int, t);
}
__device__ __forceinline__ v8f wmma_bf16(const Frag& a, const Frag& b, v8f c) {
    return __builtin_amdgcn_wmma_f32_16x16x32_bf16(
        false, a.v, false, b.v, (short)0, c, false, false);
}

// Model constants
#define CE   1408
#define CD   1024
#define CL   24
#define CNH  16
#define CB   2
#define CT   4
#define CP   256
#define CTOK 258
#define CN   1032              // T * TOK
#define CM   (CB * CN)         // 2064 rows
#define CHD  64
#define CMLP 4096
#define AINF 1e30f

// ---------------------------------------------------------------------------
// Generic bf16-WMMA GEMM:  out[M x Nc] = act( X[M x K] @ W[K x Nc] + bias ) (+res)
// block: 256 threads (8 waves), tile 128(M) x 128(N), BK = 32
// wave tile: 32(M) x 64(N)  ->  8 WMMAs per k-step per wave
// B tile is kept in LDS in *fragment order* [col16][half][lane16][p] so each
// fragment is 8 contiguous dwords -> two ds_load_b128, no register shuffles.
// K % 32 == 0, Nc % 128 == 0 (true for all call sites). M may be ragged.
// remap=1: output row r -> r + 2*(r>>8)   (embed interleave of [256 patches|2 tok])
// actmode: 0 = none, 1 = exact GELU
// ---------------------------------------------------------------------------
__global__ void __launch_bounds__(256)
gemm_bf16_kernel(const float* __restrict__ X, const float* __restrict__ W,
                 const float* __restrict__ bias, const float* __restrict__ res,
                 float* __restrict__ out,
                 int M, int K, int Nc, int actmode, int remap)
{
    __shared__ __align__(16) unsigned short lA[128 * 32];      // [row][k] bf16  (8 KB)
    __shared__ __align__(16) unsigned int   lBf[8 * 2 * 16 * 8]; // frag order   (8 KB)

    const int tid  = threadIdx.x;
    const int lane = tid & 31;
    const int wave = tid >> 5;
    const int wm   = wave & 3;        // 4 row-waves  (32 rows each)
    const int wn   = wave >> 2;       // 2 col-waves  (64 cols each)
    const int half = lane >> 4;
    const int l16  = lane & 15;

    const int bM = blockIdx.x * 128;
    const int bN = blockIdx.y * 128;

    // staging coordinates
    const int ar0 = tid >> 3;               // A row for i=0 (0..31)
    const int akq = (tid & 7) << 2;         // A k-offset 0..28
    const int bkp = tid >> 5;               // B k-pair for i=0 (0..7)
    const int bcg = (tid & 31) << 2;        // B col group 0..124

    v8f zero = {0.f,0.f,0.f,0.f,0.f,0.f,0.f,0.f};
    v8f acc[2][4] = {{zero, zero, zero, zero}, {zero, zero, zero, zero}};

    for (int k0 = 0; k0 < K; k0 += 32) {
        __syncthreads();
        // ---- stage A tile (128x32): float4 loads, clamped rows (garbage rows
        //      are discarded by the epilogue row guard)
        #pragma unroll
        for (int i = 0; i < 4; ++i) {
            int r  = ar0 + i * 32;
            int gr = bM + r; if (gr > M - 1) gr = M - 1;
            const float4 xv = *reinterpret_cast<const float4*>(
                &X[(size_t)gr * K + (k0 + akq)]);
            uint2 st; st.x = pack2(xv.x, xv.y); st.y = pack2(xv.z, xv.w);
            *reinterpret_cast<uint2*>(&lA[r * 32 + akq]) = st;
        }
        // ---- stage B tile (32x128) into fragment-order layout
        #pragma unroll
        for (int i = 0; i < 2; ++i) {
            int kp    = bkp + i * 8;
            int khalf = kp >> 3;
            int kpp   = kp & 7;
            const float* w0 = &W[(size_t)(k0 + (kp << 1)) * Nc + bN + bcg];
            const float4 a0 = *reinterpret_cast<const float4*>(w0);
            const float4 a1 = *reinterpret_cast<const float4*>(w0 + Nc);
            unsigned int pk[4];
            pk[0] = pack2(a0.x, a1.x);
            pk[1] = pack2(a0.y, a1.y);
            pk[2] = pack2(a0.z, a1.z);
            pk[3] = pack2(a0.w, a1.w);
            #pragma unroll
            for (int ci = 0; ci < 4; ++ci) {
                int col = bcg + ci;
                lBf[((((col >> 4) * 2 + khalf) * 16 + (col & 15)) << 3) + kpp] = pk[ci];
            }
        }
        if (k0 + 32 < K)
            __builtin_prefetch(&W[(size_t)(k0 + 32 + (bkp << 1)) * Nc + bN + bcg], 0, 1);
        __syncthreads();

        // ---- A fragments (2 m-subtiles): contiguous K order k = 16*half + s, same as B
        Frag af[2];
        const unsigned int* pA = (const unsigned int*)lA;
        #pragma unroll
        for (int mt = 0; mt < 2; ++mt)
            #pragma unroll
            for (int p = 0; p < 8; ++p) {
                int kk = 2 * p + 16 * half;
                af[mt].u[p] = pA[((wm * 32 + mt * 16 + l16) * 32 + kk) >> 1];
            }
        // ---- B fragments (4 n-subtiles): contiguous 8 dwords each
        Frag bf[4];
        #pragma unroll
        for (int nt = 0; nt < 4; ++nt) {
            int base = (((wn * 4 + nt) * 2 + half) * 16 + l16) << 3;
            #pragma unroll
            for (int p = 0; p < 8; ++p)
                bf[nt].u[p] = lBf[base + p];
        }

        #pragma unroll
        for (int mt = 0; mt < 2; ++mt)
            #pragma unroll
            for (int nt = 0; nt < 4; ++nt)
                acc[mt][nt] = wmma_bf16(af[mt], bf[nt], acc[mt][nt]);
    }

    // epilogue: C/D layout -> element j is (M = j + 8*half, N = lane&15)
    #pragma unroll
    for (int mt = 0; mt < 2; ++mt)
        #pragma unroll
        for (int nt = 0; nt < 4; ++nt)
            #pragma unroll
            for (int j = 0; j < 8; ++j) {
                int m   = j + 8 * half;
                int row = bM + wm * 32 + mt * 16 + m;
                if (row >= M) continue;
                int col = bN + wn * 64 + nt * 16 + l16;
                float vv = acc[mt][nt][j];
                if (bias) vv += bias[col];
                if (actmode == 1) vv = 0.5f * vv * (1.f + erff(vv * 0.70710678118654752f));
                int orow = remap ? (row + 2 * (row >> 8)) : row;
                size_t oi = (size_t)orow * Nc + col;
                if (res) vv += res[oi];
                out[oi] = vv;
            }
}

// ---------------------------------------------------------------------------
// LayerNorm: one block per row, D = 1024 (float4 path)
// ---------------------------------------------------------------------------
__global__ void __launch_bounds__(256)
ln_kernel(const float* __restrict__ x, const float* __restrict__ g,
          const float* __restrict__ b, float* __restrict__ y, int D)
{
    __shared__ float r1[256];
    __shared__ float r2[256];
    const int tid = threadIdx.x;
    const int nv  = D >> 2;
    const float4* xr = (const float4*)(x + (size_t)blockIdx.x * D);
    float s = 0.f, sq = 0.f;
    for (int i = tid; i < nv; i += 256) {
        float4 v = xr[i];
        s  += v.x + v.y + v.z + v.w;
        sq += v.x * v.x + v.y * v.y + v.z * v.z + v.w * v.w;
    }
    r1[tid] = s; r2[tid] = sq;
    __syncthreads();
    for (int off = 128; off > 0; off >>= 1) {
        if (tid < off) { r1[tid] += r1[tid + off]; r2[tid] += r2[tid + off]; }
        __syncthreads();
    }
    float mean = r1[0] / (float)D;
    float var  = r2[0] / (float)D - mean * mean;
    float rstd = rsqrtf(var + 1e-5f);
    float4* yr = (float4*)(y + (size_t)blockIdx.x * D);
    const float4* g4 = (const float4*)g;
    const float4* b4 = (const float4*)b;
    for (int i = tid; i < nv; i += 256) {
        float4 v = xr[i], gv = g4[i], bv = b4[i], o;
        o.x = (v.x - mean) * rstd * gv.x + bv.x;
        o.y = (v.y - mean) * rstd * gv.y + bv.y;
        o.z = (v.z - mean) * rstd * gv.z + bv.z;
        o.w = (v.w - mean) * rstd * gv.w + bv.w;
        yr[i] = o;
    }
}

// ---------------------------------------------------------------------------
// Action / state token projection (K = 7), writes 2 rows per (b,t)
// ---------------------------------------------------------------------------
__global__ void __launch_bounds__(256)
tok_kernel(const float* __restrict__ actions, const float* __restrict__ states,
           const float* __restrict__ aW, const float* __restrict__ ab,
           const float* __restrict__ sW, const float* __restrict__ sb,
           float* __restrict__ x)
{
    int bt = blockIdx.x;                 // 0 .. B*T-1
    int b = bt / CT, t = bt % CT;
    const float* ar = actions + bt * 7;
    const float* sr = states  + bt * 7;
    size_t base = ((size_t)(b * CN + t * CTOK + CP)) * CD;
    for (int d = threadIdx.x; d < CD; d += 256) {
        float va = ab[d], vs = sb[d];
        #pragma unroll
        for (int i = 0; i < 7; ++i) {
            va += ar[i] * aW[i * CD + d];
            vs += sr[i] * sW[i * CD + d];
        }
        x[base + d]      = va;
        x[base + CD + d] = vs;
    }
}

// ---------------------------------------------------------------------------
// Flash attention with block-causal mask.
// grid: (ceil(N/64), B*NH), block: 128 (4 waves, 16 query rows per wave)
// q/k/v/o layout: [B, N, NH*HD] (head-major columns)
// OOB keys are handled by the mask (score -> -1e30 -> p = 0), so clamped
// loads are safe; OOB query rows are clamped and dropped at the final store.
// V tile kept in fragment-order LDS layout (8 contiguous dwords per fragment).
// ---------------------------------------------------------------------------
__global__ void __launch_bounds__(128)
attn_kernel(const float* __restrict__ q, const float* __restrict__ k,
            const float* __restrict__ v, float* __restrict__ o)
{
    __shared__ __align__(16) unsigned short lK[32 * 64];       // [key][hd] bf16
    __shared__ __align__(16) unsigned int   lVf[4 * 2 * 16 * 8]; // frag order (4 KB)
    __shared__ float          lS[4 * 16 * 32];
    __shared__ __align__(16) unsigned short lP[4 * 16 * 32];   // [row][key] bf16
    __shared__ float          lAl[4 * 16];
    __shared__ float          lL[4 * 16];

    const int tid  = threadIdx.x;
    const int lane = tid & 31;
    const int wave = tid >> 5;
    const int half = lane >> 4;
    const int l16  = lane & 15;

    const int bh = blockIdx.y;
    const int b  = bh >> 4;
    const int h  = bh & 15;
    const int qb0 = blockIdx.x * 64;

    const int qrow = qb0 + wave * 16 + l16;
    const int qcl  = (qrow < CN) ? qrow : (CN - 1);
    const int qtb  = qcl / CTOK;                 // time block of owned row

    // Q fragments (A-matrix layout), loaded once, bf16
    Frag qf[2];
    const float* qp = q + ((size_t)(b * CN + qcl) * CD) + h * CHD;
    #pragma unroll
    for (int c = 0; c < 2; ++c) {
        #pragma unroll
        for (int p = 0; p < 8; ++p) {
            int kk = 2 * p + 16 * half + 32 * c;
            const float2 f = *reinterpret_cast<const float2*>(&qp[kk]);
            qf[c].u[p] = pack2(f.x, f.y);
        }
    }

    v8f zero = {0.f,0.f,0.f,0.f,0.f,0.f,0.f,0.f};
    v8f acc[4] = {zero, zero, zero, zero};
    float mi = -AINF, li = 0.f;                 // valid in lanes 0-15

    // causal key limit (uniform per block)
    int qmax = qb0 + 63; if (qmax > CN - 1) qmax = CN - 1;
    int limit = ((qmax / CTOK) + 1) * CTOK; if (limit > CN) limit = CN;
    const int nch = (limit + 31) >> 5;

    // staging coordinates
    const int kkey = tid >> 4;                  // K: key row for i=0 (0..7)
    const int khq  = (tid & 15) << 2;           // K/V: hd group 0..60
    const int vpr  = tid >> 4;                  // V: key pair for i=0

    for (int kc = 0; kc < nch; ++kc) {
        const int kb = kc * 32;
        __syncthreads();                        // WAR on LDS tiles
        // ---- stage K chunk [32 x 64] bf16 (row-major, hd pairs contiguous)
        #pragma unroll
        for (int i = 0; i < 4; ++i) {
            int key = kkey + i * 8;
            int kg  = kb + key; if (kg > CN - 1) kg = CN - 1;
            const float4 kv = *reinterpret_cast<const float4*>(
                &k[((size_t)(b * CN + kg) * CD) + h * CHD + khq]);
            uint2 st; st.x = pack2(kv.x, kv.y); st.y = pack2(kv.z, kv.w);
            *reinterpret_cast<uint2*>(&lK[key * 64 + khq]) = st;
        }
        // ---- stage V chunk [32 x 64] into fragment-order layout
        #pragma unroll
        for (int i = 0; i < 2; ++i) {
            int pr    = vpr + i * 8;            // key pair 0..15
            int vhalf = pr >> 3;
            int vpp   = pr & 7;
            int kg0 = kb + (pr << 1); if (kg0 > CN - 1) kg0 = CN - 1;
            int kg1 = kg0 + 1;        if (kg1 > CN - 1) kg1 = CN - 1;
            const float4 v0 = *reinterpret_cast<const float4*>(
                &v[((size_t)(b * CN + kg0) * CD) + h * CHD + khq]);
            const float4 v1 = *reinterpret_cast<const float4*>(
                &v[((size_t)(b * CN + kg1) * CD) + h * CHD + khq]);
            unsigned int pk[4];
            pk[0] = pack2(v0.x, v1.x);
            pk[1] = pack2(v0.y, v1.y);
            pk[2] = pack2(v0.z, v1.z);
            pk[3] = pack2(v0.w, v1.w);
            #pragma unroll
            for (int ci = 0; ci < 4; ++ci) {
                int hd = khq + ci;
                lVf[((((hd >> 4) * 2 + vhalf) * 16 + (hd & 15)) << 3) + vpp] = pk[ci];
            }
        }
        __syncthreads();

        // ---- S = Q @ K^T  (two key sub-tiles x two hd chunks)
        const unsigned int* pK = (const unsigned int*)lK;
        v8f s[2] = {zero, zero};
        #pragma unroll
        for (int t = 0; t < 2; ++t) {
            Frag bk0, bk1;
            #pragma unroll
            for (int p = 0; p < 8; ++p) {
                int key = t * 16 + l16;
                int hd0 = 2 * p + 16 * half;
                bk0.u[p] = pK[(key * 64 + hd0) >> 1];
                bk1.u[p] = pK[(key * 64 + hd0 + 32) >> 1];
            }
            s[t] = wmma_bf16(qf[0], bk0, s[t]);
            s[t] = wmma_bf16(qf[1], bk1, s[t]);
        }
        // scatter scaled scores to LDS
        #pragma unroll
        for (int t = 0; t < 2; ++t)
            #pragma unroll
            for (int j = 0; j < 8; ++j) {
                int m = j + 8 * half;
                lS[(wave * 16 + m) * 32 + t * 16 + l16] = s[t][j] * 0.125f;
            }
        __syncthreads();

        // ---- online softmax: lanes 0-15 each own one query row
        if (lane < 16) {
            float sv[32];
            float cmax = -AINF;
            #pragma unroll
            for (int kk = 0; kk < 32; ++kk) {
                int kg = kb + kk;
                float xv = lS[(wave * 16 + lane) * 32 + kk];
                bool ok = (kg < CN) && ((kg / CTOK) <= qtb);
                xv = ok ? xv : -AINF;
                sv[kk] = xv;
                cmax = fmaxf(cmax, xv);
            }
            float nm = fmaxf(mi, cmax);
            float al = expf(mi - nm);
            float ps = 0.f;
            unsigned int* lPu = (unsigned int*)lP;
            #pragma unroll
            for (int kk = 0; kk < 32; kk += 2) {
                float p0 = expf(sv[kk]     - nm);
                float p1 = expf(sv[kk + 1] - nm);
                ps += p0 + p1;
                lPu[(wave * 16 + lane) * 16 + (kk >> 1)] = pack2(p0, p1);
            }
            li = li * al + ps;
            mi = nm;
            lAl[wave * 16 + lane] = al;
        }
        __syncthreads();

        // ---- rescale O and accumulate O += P @ V
        Frag pf;
        const unsigned int* pP = (const unsigned int*)lP;
        #pragma unroll
        for (int p = 0; p < 8; ++p) {
            int kk = 2 * p + 16 * half;
            pf.u[p] = pP[((wave * 16 + l16) * 32 + kk) >> 1];
        }
        #pragma unroll
        for (int nt = 0; nt < 4; ++nt) {
            #pragma unroll
            for (int j = 0; j < 8; ++j)
                acc[nt][j] *= lAl[wave * 16 + j + 8 * half];
            Frag bv;
            int base = ((nt * 2 + half) * 16 + l16) << 3;
            #pragma unroll
            for (int p = 0; p < 8; ++p)
                bv.u[p] = lVf[base + p];
            acc[nt] = wmma_bf16(pf, bv, acc[nt]);
        }
    }

    if (lane < 16) lL[wave * 16 + lane] = (li > 0.f) ? (1.f / li) : 0.f;
    __syncthreads();

    #pragma unroll
    for (int nt = 0; nt < 4; ++nt)
        #pragma unroll
        for (int j = 0; j < 8; ++j) {
            int m = j + 8 * half;
            int r = qb0 + wave * 16 + m;
            if (r < CN) {
                float linv = lL[wave * 16 + m];
                o[((size_t)(b * CN + r) * CD) + h * CHD + nt * 16 + l16] = acc[nt][j] * linv;
            }
        }
}

// ---------------------------------------------------------------------------
// Gather last-time-block patch rows (float4): gtmp[512 x 1024] <- y rows
// ---------------------------------------------------------------------------
__global__ void __launch_bounds__(256)
gather_kernel(const float* __restrict__ y, float* __restrict__ gtmp)
{
    int idx = blockIdx.x * 256 + threadIdx.x;          // 512*256 float4s
    if (idx >= CB * CP * (CD / 4)) return;
    int d4 = idx & 255;
    int r  = idx >> 8;
    int b  = r >> 8;
    int p  = r & 255;
    int src = b * CN + (CT - 1) * CTOK + p;            // si = 774
    ((float4*)gtmp)[idx] = ((const float4*)y)[(size_t)src * 256 + d4];
}

// ---------------------------------------------------------------------------
// Broadcast prediction over predict_steps (float4): out[B,S,P,E] <- otmp[B,P,E]
// ---------------------------------------------------------------------------
__global__ void __launch_bounds__(256)
bcast_kernel(const float* __restrict__ otmp, float* __restrict__ out, int S, int total4)
{
    int idx = blockIdx.x * 256 + threadIdx.x;
    if (idx >= total4) return;
    int e4 = idx % (CE / 4);
    int r  = idx / (CE / 4);
    int p  = r % CP;
    int r2 = r / CP;
    int b  = r2 / S;
    ((float4*)out)[idx] = ((const float4*)otmp)[((size_t)(b * CP + p)) * (CE / 4) + e4];
}

// ---------------------------------------------------------------------------
// Host orchestration
// ---------------------------------------------------------------------------
static inline void launch_gemm(const float* X, const float* W, const float* bias,
                               const float* res, float* out,
                               int M, int K, int Nc, int act, int remap,
                               hipStream_t stream)
{
    dim3 grid((M + 127) / 128, Nc / 128);
    gemm_bf16_kernel<<<grid, 256, 0, stream>>>(X, W, bias, res, out, M, K, Nc, act, remap);
}

extern "C" void kernel_launch(void* const* d_in, const int* in_sizes, int n_in,
                              void* d_out, int out_size, void* d_ws, size_t ws_size,
                              hipStream_t stream)
{
    const float* encf   = (const float*)d_in[0];
    const float* acts   = (const float*)d_in[1];
    const float* stts   = (const float*)d_in[2];
    const float* enc_W  = (const float*)d_in[3];
    const float* enc_b  = (const float*)d_in[4];
    const float* act_W  = (const float*)d_in[5];
    const float* act_b  = (const float*)d_in[6];
    const float* st_W   = (const float*)d_in[7];
    const float* st_b   = (const float*)d_in[8];
    const float* ln1_g  = (const float*)d_in[9];
    const float* ln1_b  = (const float*)d_in[10];
    const float* Wq     = (const float*)d_in[11];
    const float* bq     = (const float*)d_in[12];
    const float* Wk     = (const float*)d_in[13];
    const float* bk     = (const float*)d_in[14];
    const float* Wv     = (const float*)d_in[15];
    const float* bv     = (const float*)d_in[16];
    const float* Wo     = (const float*)d_in[17];
    const float* bo     = (const float*)d_in[18];
    const float* ln2_g  = (const float*)d_in[19];
    const float* ln2_b  = (const float*)d_in[20];
    const float* W1     = (const float*)d_in[21];
    const float* b1     = (const float*)d_in[22];
    const float* W2     = (const float*)d_in[23];
    const float* b2     = (const float*)d_in[24];
    const float* nf_g   = (const float*)d_in[25];
    const float* nf_b   = (const float*)d_in[26];
    const float* out_W  = (const float*)d_in[27];
    const float* out_b  = (const float*)d_in[28];
    (void)n_in; (void)in_sizes;

    // workspace layout (floats): x, y, q, k, v, o ; MLP hidden aliases q..o
    float* ws = (float*)d_ws;
    const size_t SZ = (size_t)CM * CD;          // 2064 * 1024
    float* x   = ws;
    float* y   = x + SZ;
    float* qb_ = y + SZ;
    float* kb_ = qb_ + SZ;
    float* vb_ = kb_ + SZ;
    float* ob_ = vb_ + SZ;
    float* h1   = qb_;                          // 2064 x 4096 == 4 * SZ (q..o)
    float* gtmp = qb_;                          // 512 x 1024  (post-attention reuse)
    float* otmp = kb_;                          // 512 x 1408
    (void)ws_size;

    // ---- embed: patches GEMM with interleave remap, then action/state tokens
    launch_gemm(encf, enc_W, enc_b, nullptr, x, CB * CT * CP, CE, CD, 0, 1, stream);
    tok_kernel<<<CB * CT, 256, 0, stream>>>(acts, stts, act_W, act_b, st_W, st_b, x);

    // ---- transformer layers
    for (int l = 0; l < CL; ++l) {
        const float* g1  = ln1_g + (size_t)l * CD;
        const float* bb1 = ln1_b + (size_t)l * CD;
        const float* wq  = Wq + (size_t)l * CD * CD;
        const float* bql = bq + (size_t)l * CD;
        const float* wk  = Wk + (size_t)l * CD * CD;
        const float* bkl = bk + (size_t)l * CD;
        const float* wv  = Wv + (size_t)l * CD * CD;
        const float* bvl = bv + (size_t)l * CD;
        const float* wo  = Wo + (size_t)l * CD * CD;
        const float* bol = bo + (size_t)l * CD;
        const float* g2  = ln2_g + (size_t)l * CD;
        const float* bb2 = ln2_b + (size_t)l * CD;
        const float* w1  = W1 + (size_t)l * CD * CMLP;
        const float* b1m = b1 + (size_t)l * CMLP;
        const float* w2  = W2 + (size_t)l * CMLP * CD;
        const float* b2m = b2 + (size_t)l * CD;

        ln_kernel<<<CM, 256, 0, stream>>>(x, g1, bb1, y, CD);
        launch_gemm(y, wq, bql, nullptr, qb_, CM, CD, CD, 0, 0, stream);
        launch_gemm(y, wk, bkl, nullptr, kb_, CM, CD, CD, 0, 0, stream);
        launch_gemm(y, wv, bvl, nullptr, vb_, CM, CD, CD, 0, 0, stream);

        dim3 agrid((CN + 63) / 64, CB * CNH);
        attn_kernel<<<agrid, 128, 0, stream>>>(qb_, kb_, vb_, ob_);

        launch_gemm(ob_, wo, bol, x, x, CM, CD, CD, 0, 0, stream);        // +residual
        ln_kernel<<<CM, 256, 0, stream>>>(x, g2, bb2, y, CD);
        launch_gemm(y, w1, b1m, nullptr, h1, CM, CD, CMLP, 1, 0, stream); // GELU
        launch_gemm(h1, w2, b2m, x, x, CM, CMLP, CD, 0, 0, stream);       // +residual
    }

    // ---- final norm + output projection + broadcast
    ln_kernel<<<CM, 256, 0, stream>>>(x, nf_g, nf_b, y, CD);
    gather_kernel<<<(CB * CP * (CD / 4) + 255) / 256, 256, 0, stream>>>(y, gtmp);
    launch_gemm(gtmp, out_W, out_b, nullptr, otmp, CB * CP, CD, CE, 0, 0, stream);

    int S = out_size / (CB * CP * CE);
    if (S < 1) S = 1;
    int total4 = out_size / 4;
    bcast_kernel<<<(total4 + 255) / 256, 256, 0, stream>>>(otmp, (float*)d_out, S, total4);
}
